// WindowMSA_20890720928462
// MI455X (gfx1250) — matrix operand order repaired and checked by reading, not compiled, not run on hardware
//
#include <hip/hip_runtime.h>

// ---------------------------------------------------------------------------
// Fused Swin window-attention for gfx1250 (MI455X), wave32 + WMMA bf16.
// One workgroup (256 threads = 8 waves) handles one window end-to-end in LDS.
// Round 4: prep pass in d_ws -> bf16 weights (no f32->bf16 cvt in hot loops,
//          half the weight L2 traffic) and pre-combined rel-bias+mask table
//          (single load in the S epilogue instead of a dependent gather chain).
// ---------------------------------------------------------------------------

typedef __bf16 bf16;
typedef __attribute__((ext_vector_type(16))) __bf16 v16bf;
typedef __attribute__((ext_vector_type(8)))  float  v8f;

#define EMBED   192
#define HEADS   6
#define HD      32
#define NTOK    49
#define NPAD    64
#define NN      (NTOK * NTOK)           // 2401
#define QK_SCALE 0.17677669529663687f   // 1/sqrt(32)

// LDS carve (bytes); total 147456 < 320KB WGP LDS
#define OFF_XS   0                      // bf16 [64][192]        = 24576
#define OFF_QKV  24576                  // bf16 q[6][64][32], k[6][64][32], vt[6][32][64]
#define OFF_S    98304                  // f32  [64][64]         = 16384
#define OFF_P    114688                 // bf16 [64][64]         = 8192
#define OFF_OS   122880                 // bf16 [64][192]        = 24576
#define SMEM_BYTES 147456

#define QKV_ELEMS_PER  (HEADS * NPAD * HD)   // 12288 bf16 elements per q/k/vt

// workspace carve (bytes)
#define QKVW_ELEMS  (3 * EMBED * EMBED)      // 110592
#define PROJW_ELEMS (EMBED * EMBED)          // 36864
#define BM_ELEMS    (64 * HEADS * NN)        // 921984
#define WS_OFF_WQ   0                        // bf16[110592] = 221184 B
#define WS_OFF_WP   221184                   // bf16[36864]  =  73728 B
#define WS_OFF_BM   294912                   // f32 [921984] = 3687936 B

// ---- WMMA fragment loaders (CDNA5 ISA 7.12.2 layouts, wave32) --------------

// A-matrix 16x32 bf16: lane m=l&15, K-half kh=l>>4; elem i -> k=(i<8?i:i+8)+8*kh
__device__ __forceinline__ v16bf load_a(const bf16* base, int stride, int row0,
                                        int k0, int ln) {
  int m = ln & 15, kh = ln >> 4;
  const bf16* p = base + (row0 + m) * stride + k0 + 8 * kh;
  v16bf a;
#pragma unroll
  for (int i = 0; i < 8; ++i) a[i] = p[i];
#pragma unroll
  for (int i = 0; i < 8; ++i) a[i + 8] = p[i + 16];
  return a;
}

// B-matrix 32x16: B[k][n] = W[(n0+n)*stride + k0+k]  (column n of B = row of W)
// lane n=l&15, kh=l>>4; elem i -> k=(i<8?i:i+8)+8*kh ; bf16 source
__device__ __forceinline__ v16bf load_b_wt_bf(const bf16* w, int stride,
                                              int n0, int k0, int ln) {
  int n = ln & 15, kh = ln >> 4;
  const bf16* p = w + (n0 + n) * stride + k0 + 8 * kh;
  v16bf b;
#pragma unroll
  for (int i = 0; i < 8; ++i) b[i] = p[i];
#pragma unroll
  for (int i = 0; i < 8; ++i) b[i + 8] = p[i + 16];
  return b;
}

#define WMMA_BF16(A, B, C) \
  __builtin_amdgcn_wmma_f32_16x16x32_bf16(false, (A), false, (B), (short)0, (C), false, false)

// ---------------------------------------------------------------------------
// prep kernels (run once per launch; outputs are L2-resident for main kernel)

__global__ void __launch_bounds__(256)
prep_weights(const float* __restrict__ qkv_w, const float* __restrict__ proj_w,
             bf16* __restrict__ wq, bf16* __restrict__ wp) {
  int i = blockIdx.x * 256 + threadIdx.x;
  if (i < QKVW_ELEMS)  wq[i] = (bf16)qkv_w[i];
  if (i < PROJW_ELEMS) wp[i] = (bf16)proj_w[i];
}

__global__ void __launch_bounds__(256)
prep_bm(const float* __restrict__ mask, const float* __restrict__ bias_table,
        const int* __restrict__ rel_index, float* __restrict__ bm) {
  int idx = blockIdx.x * 256 + threadIdx.x;       // [64][HEADS][NN]
  if (idx >= BM_ELEMS) return;
  int w   = idx / (HEADS * NN);
  int rem = idx % (HEADS * NN);
  int h   = rem / NN;
  int ij  = rem % NN;
  bm[idx] = bias_table[rel_index[ij] * HEADS + h] + mask[w * NN + ij];
}

// ---------------------------------------------------------------------------

__global__ void __launch_bounds__(256)
window_msa_fused(const float* __restrict__ x,
                 const bf16*  __restrict__ wq_bf,   // [576][192] bf16
                 const float* __restrict__ qkv_b,
                 const bf16*  __restrict__ wp_bf,   // [192][192] bf16
                 const float* __restrict__ proj_b,
                 const float* __restrict__ bm,      // [64][6][49][49] f32
                 float* __restrict__ out) {
  extern __shared__ char smem[];
  bf16*  Xs   = (bf16*)(smem + OFF_XS);    // [64][192] bf16, rows 49..63 zero
  bf16*  Qs   = (bf16*)(smem + OFF_QKV);                        // [6][64][32]
  bf16*  Ks   = Qs + QKV_ELEMS_PER;                             // [6][64][32]
  bf16*  Vt   = Qs + 2 * QKV_ELEMS_PER;                         // [6][32][64]
  float* S    = (float*)(smem + OFF_S);    // [64][64] logits
  bf16*  P    = (bf16*)(smem + OFF_P);     // [64][64] probs
  bf16*  Os   = (bf16*)(smem + OFF_OS);    // [64][192] attn out (head-concat)

  const int b   = blockIdx.x;
  const int tid = threadIdx.x;
  // wave id: provably wave-uniform -> pin to SGPR so all tile routing is scalar
  const int wv  = __builtin_amdgcn_readfirstlane(tid >> 5);   // 0..7
  const int ln  = tid & 31;
  const int n   = ln & 15;                 // WMMA D: column within tile
  const int hl  = ln >> 4;                 // WMMA D: row-half select

  // ---- stage x -> LDS bf16, zero-pad rows 49..63 ----
  const float* xb = x + (long)b * (NTOK * EMBED);
  for (int i = tid; i < NPAD * EMBED; i += 256) {
    float v = (i < NTOK * EMBED) ? xb[i] : 0.0f;
    Xs[i] = (bf16)v;
  }
  __syncthreads();

  // ---- QKV projection: N-tile-major, 4 M-tile accumulators per wave ----
  for (int nt = wv; nt < 36; nt += 8) {
    v8f acc0 = {}, acc1 = {}, acc2 = {}, acc3 = {};
    for (int ks = 0; ks < 6; ++ks) {
      v16bf bm_ = load_b_wt_bf(wq_bf, EMBED, nt * 16, ks * 32, ln);
      v16bf a0 = load_a(Xs, EMBED, 0,  ks * 32, ln);
      v16bf a1 = load_a(Xs, EMBED, 16, ks * 32, ln);
      v16bf a2 = load_a(Xs, EMBED, 32, ks * 32, ln);
      v16bf a3 = load_a(Xs, EMBED, 48, ks * 32, ln);
      acc0 = WMMA_BF16(a0, bm_, acc0);
      acc1 = WMMA_BF16(a1, bm_, acc1);
      acc2 = WMMA_BF16(a2, bm_, acc2);
      acc3 = WMMA_BF16(a3, bm_, acc3);
    }
    // tile routing: all scalar (nt is SGPR) -> uniform s_cbranch, no exec masks
    const int colbase = nt * 16;
    const int which   = colbase / EMBED;          // 0=q 1=k 2=v (uniform)
    const int rembase = colbase % EMBED;
    const int hh      = rembase / HD;             // head (uniform)
    const int dd0     = rembase % HD;             // 0 or 16 (uniform)
    const int dd      = dd0 + n;
    const float bias  = qkv_b[colbase + n];

    if (which == 2) {                             // V -> transposed layout
      bf16* p0 = Vt + (hh * HD + dd) * NPAD + 8 * hl;
      auto postV = [&](v8f acc, int mt) {
        bf16* p = p0 + mt * 16;
        if (mt < 3) {
#pragma unroll
          for (int r = 0; r < 8; ++r) p[r] = (bf16)(acc[r] + bias);
        } else {
#pragma unroll
          for (int r = 0; r < 8; ++r) {
            int row = 48 + 8 * hl + r;
            p[r] = (bf16)((row < NTOK) ? (acc[r] + bias) : 0.0f);
          }
        }
      };
      postV(acc0, 0); postV(acc1, 1); postV(acc2, 2); postV(acc3, 3);
    } else {
      bf16* basep = (which == 0) ? Qs : Ks;
      const float scl = (which == 0) ? QK_SCALE : 1.0f;
      bf16* p0 = basep + (hh * NPAD + 8 * hl) * HD + dd;
      auto postQK = [&](v8f acc, int mt) {
        bf16* p = p0 + mt * 16 * HD;
        if (mt < 3) {
#pragma unroll
          for (int r = 0; r < 8; ++r) p[r * HD] = (bf16)((acc[r] + bias) * scl);
        } else {
#pragma unroll
          for (int r = 0; r < 8; ++r) {
            int row = 48 + 8 * hl + r;
            float y = (row < NTOK) ? (acc[r] + bias) : 0.0f;
            p[r * HD] = (bf16)(y * scl);
          }
        }
      };
      postQK(acc0, 0); postQK(acc1, 1); postQK(acc2, 2); postQK(acc3, 3);
    }
  }
  __syncthreads();

  // ---- attention per head ----
  const float* bmw = bm + (long)(b & 63) * (HEADS * NN);   // this window's table
  for (int h = 0; h < HEADS; ++h) {
    const bf16*  qh  = Qs + h * NPAD * HD;
    const bf16*  khp = Ks + h * NPAD * HD;
    const bf16*  vth = Vt + h * HD * NPAD;
    const float* bmh = bmw + h * NN;

    // S = q * k^T : wave -> (nt, mt-pair); k^T frag loaded once per 2 wmma
    {
      const int nt  = wv >> 1;
      const int mtb = (wv & 1) * 2;
      v16bf bm_ = load_b_wt_bf(khp, HD, nt * 16, 0, ln);
      const int gj  = nt * 16 + n;
      const int gjc = (gj < NTOK) ? gj : (NTOK - 1);   // clamp: loads in-bounds
#pragma unroll
      for (int mi = 0; mi < 2; ++mi) {
        const int mt = mtb + mi;
        v16bf a = load_a(qh, HD, mt * 16, 0, ln);
        v8f acc = {};
        acc = WMMA_BF16(a, bm_, acc);
#pragma unroll
        for (int r = 0; r < 8; ++r) {
          int gi  = mt * 16 + 8 * hl + r;
          int gic = (gi < NTOK) ? gi : (NTOK - 1);
          float add = bmh[gic * NTOK + gjc];           // pre-combined bias+mask
          bool valid = (gi < NTOK) && (gj < NTOK);
          S[gi * NPAD + gj] = valid ? (acc[r] + add) : -1e30f;
        }
      }
    }
    __syncthreads();

    // softmax: 4 lanes per row (adjacent lanes), shfl_xor reductions
    {
      const int rowi = tid >> 2;            // 0..63
      const int quad = tid & 3;             // 0..3
      const float* row = S + rowi * NPAD + quad * 16;
      float m = -3.4e38f;
#pragma unroll
      for (int j = 0; j < 16; ++j) m = fmaxf(m, row[j]);
      m = fmaxf(m, __shfl_xor(m, 1, 32));
      m = fmaxf(m, __shfl_xor(m, 2, 32));
      float e[16];
      float sum = 0.0f;
#pragma unroll
      for (int j = 0; j < 16; ++j) { e[j] = __expf(row[j] - m); sum += e[j]; }
      sum += __shfl_xor(sum, 1, 32);
      sum += __shfl_xor(sum, 2, 32);
      const float inv = 1.0f / sum;
      unsigned* prow32 = (unsigned*)(P + rowi * NPAD + quad * 16);
#pragma unroll
      for (int j = 0; j < 8; ++j) {         // packed b32 stores
        union { bf16 h2[2]; unsigned u; } pk;
        pk.h2[0] = (bf16)(e[2 * j] * inv);
        pk.h2[1] = (bf16)(e[2 * j + 1] * inv);
        prow32[j] = pk.u;
      }
    }
    __syncthreads();

    // O = P * V : one 16x16 tile per wave; B-frags contiguous via Vt
    {
      const int mt = wv >> 1, ntv = wv & 1;
      v8f acc = {};
#pragma unroll
      for (int ks = 0; ks < 2; ++ks) {
        v16bf a   = load_a(P, NPAD, mt * 16, ks * 32, ln);
        v16bf bm_ = load_b_wt_bf(vth, NPAD, ntv * 16, ks * 32, ln);
        acc = WMMA_BF16(a, bm_, acc);
      }
      bf16* p = Os + (mt * 16 + 8 * hl) * EMBED + h * HD + ntv * 16 + n;
#pragma unroll
      for (int r = 0; r < 8; ++r) p[r * EMBED] = (bf16)acc[r];
    }
    __syncthreads();   // protect S/P reuse next head
  }

  // ---- output projection: N-tile-major, 2 M-tile accumulators per unit ----
  float* ob = out + (long)b * (NTOK * EMBED);
  for (int u = wv; u < 24; u += 8) {
    const int nt = u >> 1, mh = u & 1;
    v8f acc0 = {}, acc1 = {};
    for (int ks = 0; ks < 6; ++ks) {
      v16bf bm_ = load_b_wt_bf(wp_bf, EMBED, nt * 16, ks * 32, ln);
      v16bf a0 = load_a(Os, EMBED, (mh * 2) * 16,     ks * 32, ln);
      v16bf a1 = load_a(Os, EMBED, (mh * 2 + 1) * 16, ks * 32, ln);
      acc0 = WMMA_BF16(a0, bm_, acc0);
      acc1 = WMMA_BF16(a1, bm_, acc1);
    }
    const int col  = nt * 16 + n;
    const float pb = proj_b[col];
    auto store = [&](v8f acc, int mt) {
      float* p = ob + (mt * 16 + 8 * hl) * EMBED + col;
      if (mt < 3) {                           // rows provably < 49: no guard
#pragma unroll
        for (int r = 0; r < 8; ++r) p[r * EMBED] = acc[r] + pb;
      } else {
#pragma unroll
        for (int r = 0; r < 8; ++r) {
          int row = 48 + 8 * hl + r;
          if (row < NTOK) p[r * EMBED] = acc[r] + pb;
        }
      }
    };
    store(acc0, mh * 2); store(acc1, mh * 2 + 1);
  }
}

// ---------------------------------------------------------------------------

extern "C" void kernel_launch(void* const* d_in, const int* in_sizes, int n_in,
                              void* d_out, int out_size, void* d_ws, size_t ws_size,
                              hipStream_t stream) {
  const float* x          = (const float*)d_in[0];
  const float* mask       = (const float*)d_in[1];
  const float* qkv_w      = (const float*)d_in[2];
  const float* qkv_b      = (const float*)d_in[3];
  const float* proj_w     = (const float*)d_in[4];
  const float* proj_b     = (const float*)d_in[5];
  const float* bias_table = (const float*)d_in[6];
  const int*   rel_index  = (const int*)d_in[7];
  float* out = (float*)d_out;

  char* ws = (char*)d_ws;
  bf16*  wq_bf = (bf16*)(ws + WS_OFF_WQ);
  bf16*  wp_bf = (bf16*)(ws + WS_OFF_WP);
  float* bm    = (float*)(ws + WS_OFF_BM);

  const int B = in_sizes[0] / (NTOK * EMBED);   // 2048

  // prep: bf16 weights + combined rel-bias/mask table (tiny, L2-resident)
  prep_weights<<<(QKVW_ELEMS + 255) / 256, 256, 0, stream>>>(qkv_w, proj_w,
                                                             wq_bf, wp_bf);
  prep_bm<<<(BM_ELEMS + 255) / 256, 256, 0, stream>>>(mask, bias_table,
                                                      rel_index, bm);

  (void)hipFuncSetAttribute((const void*)window_msa_fused,
                            hipFuncAttributeMaxDynamicSharedMemorySize,
                            SMEM_BYTES);

  window_msa_fused<<<B, 256, SMEM_BYTES, stream>>>(
      x, wq_bf, qkv_b, wp_bf, proj_b, bm, out);
}
